// Word2Vec_30691836297409
// MI455X (gfx1250) — matrix operand order repaired and checked by reading, not compiled, not run on hardware
//
#include <hip/hip_runtime.h>

// CBOW word2vec loss, fused GEMM + online log-softmax for MI455X (gfx1250).
// Logits [4096 x 50257] are never materialized. Each block owns 32 batch rows
// (2 WMMA M-tiles per wave -> 32 FLOP per loaded B-byte), 8 waves stride the
// vocab with scalar (readfirstlane) loop control. B fragments are software-
// pipelined: sched_barriers pin "issue loads for t+8 -> compute t -> commit",
// so the loadcnt wait lands after the v_wmma chain, not before it.

typedef __attribute__((ext_vector_type(16))) __bf16 v16bf;
typedef __attribute__((ext_vector_type(8)))  float  v8f;

#define VOCAB  50257
#define VPAD   50272           // next multiple of 16
#define DM     128
#define BATCH  4096
#define NWIN   10
#define NTILES (VPAD / 16)     // 3142 vocab tiles of width 16
#define LIMIT  (NTILES - 1)    // tiles [0, LIMIT) need no pad masking
#define WAVES  8
#define MROWS  32              // batch rows per block (2 M-tiles)

union V16BF {
    v16bf v;
    uint4 q[2];
};

// ---------------------------------------------------------------------------
// Kernel 1: cbow[b][d] = mean_j emb[ctx[b][j]][d], stored as bf16.
// ---------------------------------------------------------------------------
__global__ void cbow_kernel(const int* __restrict__ ctx,
                            const float* __restrict__ emb,
                            __bf16* __restrict__ cb) {
    const int b = blockIdx.x;
    const int d = threadIdx.x;
    const int* c = ctx + (size_t)b * NWIN;
    float s = 0.f;
#pragma unroll
    for (int j = 0; j < NWIN; ++j)
        s += emb[(size_t)c[j] * DM + d];
    cb[(size_t)b * DM + d] = (__bf16)(s * (1.0f / NWIN));
}

// ---------------------------------------------------------------------------
// Kernel 2: W (f32) -> Wb (bf16), zero-padded to VPAD rows.
// ---------------------------------------------------------------------------
__global__ void wconv_kernel(const float* __restrict__ W,
                             __bf16* __restrict__ Wb) {
    const size_t i = (size_t)blockIdx.x * 256 + threadIdx.x;
    if (i >= (size_t)VPAD * DM) return;
    const float x = (i < (size_t)VOCAB * DM) ? W[i] : 0.f;
    Wb[i] = (__bf16)x;
}

// ---------------------------------------------------------------------------
// Fused-kernel helpers
// ---------------------------------------------------------------------------
__device__ __forceinline__ v8f wmma_bf16(v16bf a, v16bf b, v8f c) {
    return __builtin_amdgcn_wmma_f32_16x16x32_bf16(
        false, a, false, b, (short)0, c, false, false);
}

__device__ __forceinline__ void online(float x, float& m, float& s) {
    const float mn = fmaxf(m, x);
    s = s * __expf(m - mn) + __expf(x - mn);
    m = mn;
}

// B fragment: 16-bit B 32x16 layout -- lane = column N, K pairs per VGPR;
// lanes 0-15 hold K=kb*32+{0..7,16..23}, lanes 16-31 hold K=kb*32+{8..15,24..31}.
// B[k][n] = Wb[n0+n][k] -> two 16B loads per lane per kb.
__device__ __forceinline__ void load_btile(uint4* q, const __bf16* Wb,
                                           int n0, int m16, int hi) {
    const __bf16* brow = Wb + (size_t)(n0 + m16) * DM + hi * 8;
#pragma unroll
    for (int kb = 0; kb < 4; ++kb) {
        const uint4* p = reinterpret_cast<const uint4*>(brow + kb * 32);
        q[2 * kb]     = p[0];
        q[2 * kb + 1] = p[2];
    }
}

__device__ __forceinline__ void prefetch_btile(const __bf16* Wb,
                                               int n0, int m16, int hi) {
    const __bf16* brow = Wb + (size_t)(n0 + m16) * DM + hi * 16;
    __builtin_prefetch(brow, 0, 1);          // -> global_prefetch_b8
    __builtin_prefetch(brow + 64, 0, 1);
}

template <bool MASKED>
__device__ __forceinline__ void compute_tile(const V16BF a0[4], const V16BF a1[4],
                                             const uint4 q[8], int n0, int m16,
                                             float* m0, float* s0,
                                             float* m1, float* s1) {
    v8f c0 = {}, c1 = {};
#pragma unroll
    for (int kb = 0; kb < 4; ++kb) {
        V16BF bf;
        bf.q[0] = q[2 * kb];
        bf.q[1] = q[2 * kb + 1];
        c0 = wmma_bf16(a0[kb].v, bf.v, c0);
        c1 = wmma_bf16(a1[kb].v, bf.v, c1);
    }
    const bool valid = !MASKED || ((n0 + m16) < VOCAB);
#pragma unroll
    for (int r = 0; r < 8; ++r) {
        online(valid ? c0[r] : -1e30f, m0[r], s0[r]);
        online(valid ? c1[r] : -1e30f, m1[r], s1[r]);
    }
}

// ---------------------------------------------------------------------------
// Kernel 3: fused logits-GEMM + online softmax + NLL.
// Block = 256 threads (8 waves), owns batch rows [32*blockIdx.x, +32).
// ---------------------------------------------------------------------------
__global__ __launch_bounds__(256) void fused_kernel(
        const __bf16* __restrict__ cb,
        const __bf16* __restrict__ Wb,
        const int* __restrict__ center,
        float* __restrict__ loss) {
    __shared__ float lm[WAVES][MROWS];
    __shared__ float ls[WAVES][MROWS];
    __shared__ float lnll[MROWS];

    const int tid   = threadIdx.x;
    const int wave  = tid >> 5;
    const int lane  = tid & 31;
    const int m16   = lane & 15;   // A row within tile / B,C column N
    const int hi    = lane >> 4;   // half-wave selector
    const int bbase = blockIdx.x * MROWS;

    // Wave id is uniform; force it into an SGPR so all vocab-loop control is
    // SALU (s_cmp/s_cbranch) and B base addresses are SGPR-carried.
    const int wv = __builtin_amdgcn_readfirstlane(wave);

    // --- A fragments: 16-bit A 16x32 layout (ISA 7.12.2), 2 M-tiles ---------
    // lane<16 : row M=lane,    K = kb*32 + {0..7, 16..23}
    // lane>=16: row M=lane-16, K = kb*32 + {8..15, 24..31}
    V16BF a0[4], a1[4];
    {
        const __bf16* r0 = cb + (size_t)(bbase + m16) * DM;
        const __bf16* r1 = r0 + (size_t)16 * DM;
#pragma unroll
        for (int kb = 0; kb < 4; ++kb) {
            a0[kb].q[0] = *reinterpret_cast<const uint4*>(r0 + kb * 32 + hi * 8);
            a0[kb].q[1] = *reinterpret_cast<const uint4*>(r0 + kb * 32 + 16 + hi * 8);
            a1[kb].q[0] = *reinterpret_cast<const uint4*>(r1 + kb * 32 + hi * 8);
            a1[kb].q[1] = *reinterpret_cast<const uint4*>(r1 + kb * 32 + 16 + hi * 8);
        }
    }

    // Online-softmax state: C VGPR r holds row (hi*8 + r) [+16 for tile 1].
    float m0[8], s0[8], m1[8], s1[8];
#pragma unroll
    for (int r = 0; r < 8; ++r) {
        m0[r] = m1[r] = -1e30f;
        s0[r] = s1[r] = 0.f;
    }

    // --- Vocab sweep over unmasked tiles [wv, LIMIT) step 8 ------------------
    // Software pipeline, schedule pinned with sched_barriers:
    //   [loads t+8 | prefetch t+16] || [8x WMMA + softmax on t] || [commit]
    uint4 cur[8], nxt[8];
    load_btile(cur, Wb, wv * 16, m16, hi);
    for (int t = wv; t < LIMIT; t += WAVES) {
        const int tn = (t + WAVES < LIMIT) ? (t + WAVES) : t;
        load_btile(nxt, Wb, tn * 16, m16, hi);
        const int tp = (tn + WAVES < LIMIT) ? (tn + WAVES) : tn;
        prefetch_btile(Wb, tp * 16, m16, hi);
        __builtin_amdgcn_sched_barrier(0);   // loads stay above the WMMAs
        compute_tile<false>(a0, a1, cur, t * 16, m16, m0, s0, m1, s1);
        __builtin_amdgcn_sched_barrier(0);   // commit (and its wait) stays below
#pragma unroll
        for (int i = 0; i < 8; ++i) cur[i] = nxt[i];
    }

    // --- Peeled pad-masked tail tile (t = LIMIT, owned by one wave) ----------
    if ((LIMIT % WAVES) == wv) {
        load_btile(cur, Wb, LIMIT * 16, m16, hi);
        compute_tile<true>(a0, a1, cur, LIMIT * 16, m16, m0, s0, m1, s1);
    }

    // Butterfly combine across the 16 columns: xor offsets 1..8 stay inside
    // each 16-lane half, which is exactly one C-matrix row's column set.
#pragma unroll
    for (int off = 1; off < 16; off <<= 1) {
#pragma unroll
        for (int r = 0; r < 8; ++r) {
            {
                const float mo = __shfl_xor(m0[r], off, 32);
                const float so = __shfl_xor(s0[r], off, 32);
                const float mn = fmaxf(m0[r], mo);
                s0[r] = s0[r] * __expf(m0[r] - mn) + so * __expf(mo - mn);
                m0[r] = mn;
            }
            {
                const float mo = __shfl_xor(m1[r], off, 32);
                const float so = __shfl_xor(s1[r], off, 32);
                const float mn = fmaxf(m1[r], mo);
                s1[r] = s1[r] * __expf(m1[r] - mn) + so * __expf(mo - mn);
                m1[r] = mn;
            }
        }
    }

    if (m16 == 0) {            // lanes 0 and 16 carry rows 0-7 / 8-15 per tile
#pragma unroll
        for (int r = 0; r < 8; ++r) {
            lm[wave][hi * 8 + r]      = m0[r];
            ls[wave][hi * 8 + r]      = s0[r];
            lm[wave][16 + hi * 8 + r] = m1[r];
            ls[wave][16 + hi * 8 + r] = s1[r];
        }
    }
    __syncthreads();

    if (tid < MROWS) {
        float mt = lm[0][tid], st = ls[0][tid];
#pragma unroll
        for (int w = 1; w < WAVES; ++w) {
            const float mo = lm[w][tid], so = ls[w][tid];
            const float mn = fmaxf(mt, mo);
            st = st * __expf(mt - mn) + so * __expf(mo - mn);
            mt = mn;
        }
        // Center-word logit from the same bf16 operands (consistent with GEMM).
        const int b   = bbase + tid;
        const int cid = center[b];
        const __bf16* x = cb + (size_t)b   * DM;
        const __bf16* w = Wb + (size_t)cid * DM;
        float dot = 0.f;
#pragma unroll 8
        for (int d = 0; d < DM; ++d)
            dot += (float)x[d] * (float)w[d];
        lnll[tid] = mt + __logf(st) - dot;   // -log softmax at center id
    }
    __syncthreads();

    if (tid == 0) {
        float s = 0.f;
#pragma unroll
        for (int r = 0; r < MROWS; ++r) s += lnll[r];
        atomicAdd(loss, s * (1.0f / BATCH));
    }
}

// ---------------------------------------------------------------------------
extern "C" void kernel_launch(void* const* d_in, const int* in_sizes, int n_in,
                              void* d_out, int out_size, void* d_ws, size_t ws_size,
                              hipStream_t stream) {
    const int*   ctx = (const int*)d_in[0];    // [4096, 10]
    const int*   cen = (const int*)d_in[1];    // [4096]
    const float* emb = (const float*)d_in[2];  // [50257, 128]
    const float* W   = (const float*)d_in[3];  // [50257, 128]
    float* out = (float*)d_out;                // scalar loss

    __bf16* cb = (__bf16*)d_ws;                                        // 1 MB
    __bf16* Wb = (__bf16*)((char*)d_ws + (size_t)BATCH * DM * 2);      // 12.9 MB

    cbow_kernel<<<BATCH, DM, 0, stream>>>(ctx, emb, cb);

    const int welems = VPAD * DM;
    wconv_kernel<<<(welems + 255) / 256, 256, 0, stream>>>(W, Wb);

    hipMemsetAsync(out, 0, sizeof(float), stream);

    fused_kernel<<<BATCH / MROWS, 256, 0, stream>>>(cb, Wb, cen, out);
}
